// GAT_layer__26259430048709
// MI455X (gfx1250) — matrix operand order repaired and checked by reading, not compiled, not run on hardware
//
#include <hip/hip_runtime.h>
#include <hip/hip_bf16.h>

typedef __attribute__((ext_vector_type(16))) _Float16 v16h;
typedef __attribute__((ext_vector_type(8)))  float    v8f;

static constexpr int kN   = 1024;  // nodes
static constexpr int kDIN = 128;   // input features
static constexpr int kP   = 64;    // projected dim
static constexpr int kE   = 32;    // edge features (= WMMA K)
static constexpr int kH   = 160;   // attention hidden (2P+E)
static constexpr int kNT  = 10;    // H tiles of 16
static constexpr int kCH  = 64;    // j-chunk per block iteration (16 per wave)
static constexpr int kPS  = 164;   // padded LDS row stride: 16B-aligned rows,
                                   // lane start banks = multiples of 4 -> b128 conflict-free

__device__ __forceinline__ unsigned int pk2(float a, float b) {
  union { _Float16 h[2]; unsigned int u; } t;
  t.h[0] = (_Float16)a; t.h[1] = (_Float16)b; return t.u;
}

// Branch-free exact-form gelu: 0.5x(1+erf(x/sqrt2)) with A&S 7.1.26 erf
// (max abs err 1.5e-7; straight-line: 1 rcp + 1 exp2 + ~11 VALU, no EXEC branches)
__device__ __forceinline__ float gelu_fast(float x) {
  const float z  = 0.70710678118654752440f * x;
  const float az = fabsf(z);
  const float t  = __builtin_amdgcn_rcpf(fmaf(0.3275911f, az, 1.0f));
  float p = fmaf(t, 1.061405429f, -1.453152027f);
  p = fmaf(t, p, 1.421413741f);
  p = fmaf(t, p, -0.284496736f);
  p = fmaf(t, p, 0.254829592f);
  p = p * t;
  const float m = az * az;
  const float e = __builtin_amdgcn_exp2f(m * -1.4426950408889634f); // exp(-z^2)
  float er = fmaf(-p, e, 1.0f);                                     // erf(|z|)
  er = copysignf(er, x);
  const float hx = 0.5f * x;
  return fmaf(hx, er, hx);
}

// ---------------------------------------------------------------------------
// Kernel 1: h = LN1(x) @ W1 + b1 ; hiB = h@Wi + ba1 ; hj = h@Wj
// ---------------------------------------------------------------------------
__global__ __launch_bounds__(128)
void gat_prep(const float* __restrict__ x,
              const float* __restrict__ ln1g, const float* __restrict__ ln1b,
              const float* __restrict__ W1,   const float* __restrict__ b1,
              const float* __restrict__ Wa1,  const float* __restrict__ ba1,
              float* __restrict__ hiB, float* __restrict__ hjw,
              float* __restrict__ hbuf)
{
  const int i = blockIdx.x, tid = threadIdx.x;
  __shared__ float sX[kDIN];
  __shared__ float sH[kP];
  __shared__ float sRed[128];

  float v = x[(size_t)i * kDIN + tid];
  sRed[tid] = v; __syncthreads();
  for (int s = 64; s > 0; s >>= 1) { if (tid < s) sRed[tid] += sRed[tid + s]; __syncthreads(); }
  const float mean = sRed[0] * (1.0f / kDIN); __syncthreads();
  const float dv = v - mean;
  sRed[tid] = dv * dv; __syncthreads();
  for (int s = 64; s > 0; s >>= 1) { if (tid < s) sRed[tid] += sRed[tid + s]; __syncthreads(); }
  const float var = sRed[0] * (1.0f / kDIN); __syncthreads();
  sX[tid] = dv * rsqrtf(var + 1e-5f) * ln1g[tid] + ln1b[tid];
  __syncthreads();

  // h = xn @ W1 + b1  (128 threads: column = tid&63, k-half = tid>>6)
  const int c = tid & 63, part = tid >> 6;
  float a = 0.f;
  for (int k = part * 64; k < part * 64 + 64; ++k) a = fmaf(sX[k], W1[k * kP + c], a);
  sRed[tid] = a; __syncthreads();
  if (tid < kP) {
    float hv = sRed[tid] + sRed[tid + 64] + b1[tid];
    sH[tid] = hv;
    hbuf[(size_t)i * kP + tid] = hv;
  }
  __syncthreads();

  // hiB = h@Wi + ba1 ; hj = h@Wj   (Wi = Wa1[0:64], Wj = Wa1[64:128])
  for (int cc = tid; cc < kH; cc += 128) {
    float ai = 0.f, aj = 0.f;
    for (int k = 0; k < kP; ++k) {
      float hk = sH[k];
      ai = fmaf(hk, Wa1[k * kH + cc], ai);
      aj = fmaf(hk, Wa1[(kP + k) * kH + cc], aj);
    }
    hiB[(size_t)i * kH + cc] = ai + ba1[cc];
    hjw[(size_t)i * kH + cc] = aj;
  }
}

// ---------------------------------------------------------------------------
// Kernel 2 (fused): per row i — edge GEMM via WMMA, gelu, score, softmax,
// att@h, LN2. One block (4 waves, 128 threads) per i.
// WMMA orientation: D[M=hcol, N=pair], A = We^T tile (f16), B = ef tile (f16).
// ---------------------------------------------------------------------------
__global__ __launch_bounds__(128)
void gat_main(const float* __restrict__ ef,  const float* __restrict__ adj,
              const float* __restrict__ Wa1, const float* __restrict__ Wa2,
              const float* __restrict__ ba2,
              const float* __restrict__ ln2g, const float* __restrict__ ln2b,
              const float* __restrict__ hiB,  const float* __restrict__ hjw,
              const float* __restrict__ hbuf, float* __restrict__ out)
{
  const int i = blockIdx.x, tid = threadIdx.x;
  const int lane = tid & 31, wave = tid >> 5;
  const int p  = lane & 15;        // pair within the wave's 16-j tile (N index)
  const int hf = lane >> 4;        // lane half (controls K / M sub-ranges)
  const int half8 = hf * 8;

  __shared__ unsigned int sA[kNT * 256];   // packed f16 We^T in A-operand layout
  __shared__ float sPre[kCH * kPS];        // hiB[i,c] + hj[j,c] for current chunk
  __shared__ float sScore[kN];
  __shared__ float sHiB[kH];
  __shared__ float sW2[kH];
  __shared__ float sRed[128];
  __shared__ float sY[kP];

  for (int idx = tid; idx < kH; idx += 128) {
    sHiB[idx] = hiB[(size_t)i * kH + idx];
    sW2[idx]  = Wa2[idx];
  }
  // Pack We^T tiles into the ISA 16-bit A-matrix 16x32 layout:
  // lane<16: M=lane, V0..3 -> K=0..7, V4..7 -> K=16..23; lane>=16: K +8.
  for (int idx = tid; idx < kNT * 256; idx += 128) {
    int v  = idx & 7;
    int ln = (idx >> 3) & 31;
    int t  = idx >> 8;
    int lh = ln >> 4, m = ln & 15;
    int hcol = t * 16 + m;
    int kb = (v < 4) ? (2 * v) : (16 + 2 * (v - 4));
    int k0 = kb + lh * 8;
    float lo = Wa1[(size_t)(2 * kP + k0)     * kH + hcol];
    float hi = Wa1[(size_t)(2 * kP + k0 + 1) * kH + hcol];
    sA[idx] = pk2(lo, hi);
  }
  const float ba2v = ba2[0];

  union AU { v16h v; uint4 q[2]; };
  union BU { v16h v; unsigned int u[8]; };
  union F8 { float4 q[2]; float f[8]; };
  v8f cz = {};

  for (int chunk = 0; chunk < kN / kCH; ++chunk) {
    __syncthreads();
    const int jb = chunk * kCH;
    // Stage pre[jl, c] = hiB[i,c] + hj[jb+jl, c]  (coalesced 40KB global read)
    for (int idx = tid; idx < kCH * kH; idx += 128) {
      int jl = idx / kH, c = idx - jl * kH;
      sPre[jl * kPS + c] = hjw[(size_t)(jb + jl) * kH + c] + sHiB[c];
    }
    __syncthreads();

    const int j = jb + wave * 16 + p;
    // B operand: ef[i, j, :] -> lane<16 holds K=0..7,16..23, lane>=16 holds K=8..15,24..31
    const float4* efp = (const float4*)(ef + ((size_t)i * kN + j) * kE + hf * 8);
    float4 f0 = efp[0], f1 = efp[1], f2 = efp[4], f3 = efp[5];
    BU B;
    B.u[0] = pk2(f0.x, f0.y); B.u[1] = pk2(f0.z, f0.w);
    B.u[2] = pk2(f1.x, f1.y); B.u[3] = pk2(f1.z, f1.w);
    B.u[4] = pk2(f2.x, f2.y); B.u[5] = pk2(f2.z, f2.w);
    B.u[6] = pk2(f3.x, f3.y); B.u[7] = pk2(f3.z, f3.w);

    const float* preRow = &sPre[(wave * 16 + p) * kPS];
    float acc = 0.f;
    #pragma unroll
    for (int t = 0; t < kNT; ++t) {
      AU A;
      const uint4* ap = (const uint4*)&sA[t * 256 + lane * 8];
      A.q[0] = ap[0]; A.q[1] = ap[1];
      v8f d = __builtin_amdgcn_wmma_f32_16x16x32_f16(
          false, A.v, false, B.v, (short)0, cz, false, false);
      // vectorized LDS reads: rows 16B-aligned, start banks tile 64 banks
      F8 P, W;
      const float4* pr4 = (const float4*)(preRow + t * 16 + half8);
      const float4* w4  = (const float4*)(sW2    + t * 16 + half8);
      P.q[0] = pr4[0]; P.q[1] = pr4[1];
      W.q[0] = w4[0];  W.q[1] = w4[1];
      #pragma unroll
      for (int r = 0; r < 8; ++r) {
        float pre = d[r] + P.f[r];
        acc = fmaf(gelu_fast(pre), W.f[r], acc);
      }
    }
    acc += __shfl_xor(acc, 16, 32);            // combine the two M-halves
    if (hf == 0) {
      float s = acc + ba2v;
      s = (s >= 0.f) ? s : 0.2f * s;           // leaky relu
      bool keep = (adj[(size_t)i * kN + j] > 0.f) || (j == i);
      sScore[j] = keep ? s : -1.0e12f;
    }
  }
  __syncthreads();

  // Softmax over 1024 scores
  float m = -3.0e38f;
  for (int jj = tid; jj < kN; jj += 128) m = fmaxf(m, sScore[jj]);
  sRed[tid] = m; __syncthreads();
  for (int s = 64; s > 0; s >>= 1) { if (tid < s) sRed[tid] = fmaxf(sRed[tid], sRed[tid + s]); __syncthreads(); }
  const float M = sRed[0]; __syncthreads();
  float lsum = 0.f;
  for (int jj = tid; jj < kN; jj += 128) {
    float e = __expf(sScore[jj] - M);
    sScore[jj] = e; lsum += e;
  }
  sRed[tid] = lsum; __syncthreads();
  for (int s = 64; s > 0; s >>= 1) { if (tid < s) sRed[tid] += sRed[tid + s]; __syncthreads(); }
  const float S = sRed[0]; __syncthreads();

  // y = (att @ h)[i, :]   (column = tid&63, j-half = tid>>6)
  const int c = tid & 63, part = tid >> 6;
  float a = 0.f;
  for (int j = part * 512; j < part * 512 + 512; ++j)
    a = fmaf(sScore[j], hbuf[(size_t)j * kP + c], a);
  sRed[tid] = a; __syncthreads();
  float y = 0.f;
  if (tid < kP) { y = (sRed[tid] + sRed[tid + 64]) / S; sY[tid] = y; }
  __syncthreads();

  // LN2 over 64 values
  sRed[tid] = (tid < kP) ? sY[tid] : 0.f; __syncthreads();
  for (int s = 64; s > 0; s >>= 1) { if (tid < s) sRed[tid] += sRed[tid + s]; __syncthreads(); }
  const float mean = sRed[0] * (1.0f / kP); __syncthreads();
  const float dv = y - mean;
  sRed[tid] = (tid < kP) ? dv * dv : 0.f; __syncthreads();
  for (int s = 64; s > 0; s >>= 1) { if (tid < s) sRed[tid] += sRed[tid + s]; __syncthreads(); }
  const float var = sRed[0] * (1.0f / kP);
  if (tid < kP)
    out[(size_t)i * kP + tid] = dv * rsqrtf(var + 1e-5f) * ln2g[tid] + ln2b[tid];
}

// ---------------------------------------------------------------------------
extern "C" void kernel_launch(void* const* d_in, const int* in_sizes, int n_in,
                              void* d_out, int out_size, void* d_ws, size_t ws_size,
                              hipStream_t stream) {
  const float* x    = (const float*)d_in[0];
  const float* ef   = (const float*)d_in[1];
  const float* adj  = (const float*)d_in[2];
  const float* ln1g = (const float*)d_in[3];
  const float* ln1b = (const float*)d_in[4];
  const float* W1   = (const float*)d_in[5];
  const float* b1   = (const float*)d_in[6];
  const float* Wa1  = (const float*)d_in[7];
  const float* ba1  = (const float*)d_in[8];
  const float* Wa2  = (const float*)d_in[9];
  const float* ba2  = (const float*)d_in[10];
  const float* ln2g = (const float*)d_in[11];
  const float* ln2b = (const float*)d_in[12];
  float* out = (float*)d_out;

  float* ws   = (float*)d_ws;
  float* hiB  = ws;                            // 1024*160
  float* hjw  = ws + (size_t)kN * kH;          // 1024*160
  float* hbuf = ws + (size_t)2 * kN * kH;      // 1024*64

  gat_prep<<<kN, 128, 0, stream>>>(x, ln1g, ln1b, W1, b1, Wa1, ba1, hiB, hjw, hbuf);
  gat_main<<<kN, 128, 0, stream>>>(ef, adj, Wa1, Wa2, ba2, ln2g, ln2b,
                                   hiB, hjw, hbuf, out);
}
